// MultiHeadAttention_20066087207151
// MI455X (gfx1250) — matrix operand order repaired and checked by reading, not compiled, not run on hardware
//
#include <hip/hip_runtime.h>
#include <hip/hip_bf16.h>
#include <stdint.h>

// ---------------------------------------------------------------------------
// MI455X (gfx1250, wave32) multi-head attention, bf16 WMMA pipeline with
// async global->LDS staging (ASYNCcnt) and double-buffered GEMM slabs.
// ---------------------------------------------------------------------------

typedef __attribute__((ext_vector_type(16))) __bf16 v16bf;
typedef __attribute__((ext_vector_type(8)))  float  v8f;

__device__ __forceinline__ unsigned short f32_bf16(float f) {
    union { float f; uint32_t u; } v; v.f = f;
    uint32_t r = v.u + 0x7FFFu + ((v.u >> 16) & 1u);   // round-to-nearest-even
    return (unsigned short)(r >> 16);
}

union FragBF { uint4 q[2]; v16bf v; };

__device__ __forceinline__ v8f v8f_zero() {
    v8f z; for (int i = 0; i < 8; ++i) z[i] = 0.0f; return z;
}

__device__ __forceinline__ v8f wmma_bf16(const FragBF& a, const FragBF& b, v8f c) {
    // (neg_a, A, neg_b, B, c_mod, C, reuse_a, reuse_b)
    return __builtin_amdgcn_wmma_f32_16x16x32_bf16(false, a.v, false, b.v,
                                                   (short)0, c, false, false);
}

// Async 16B global -> LDS copy (GVS mode: 64-bit SGPR base + 32-bit VGPR off).
// Tracked by ASYNCcnt; lds_addr is an LDS byte address (low 32 bits of the
// generic shared-aperture address).
__device__ __forceinline__ void async_copy_b128(uint32_t lds_addr, uint64_t base,
                                                uint32_t goff) {
    asm volatile("global_load_async_to_lds_b128 %0, %1, %2"
                 :: "v"(lds_addr), "v"(goff), "s"(base) : "memory");
}
__device__ __forceinline__ void wait_async_le4() {
    asm volatile("s_wait_asynccnt 0x4" ::: "memory");
}
__device__ __forceinline__ void wait_async_0() {
    asm volatile("s_wait_asynccnt 0x0" ::: "memory");
}
__device__ __forceinline__ uint32_t lds_offset(const void* p) {
    return (uint32_t)(uintptr_t)p;   // shared aperture lives in the high 32 bits
}

// ---------------------------------------------------------------------------
// f32 -> bf16 conversion, 4 elements / thread
// ---------------------------------------------------------------------------
__global__ void cvt_f32_bf16_kernel(const float* __restrict__ in,
                                    unsigned short* __restrict__ out, int n4) {
    int i = blockIdx.x * blockDim.x + threadIdx.x;
    if (i < n4) {
        float4 f = ((const float4*)in)[i];
        unsigned int lo = (unsigned int)f32_bf16(f.x) | ((unsigned int)f32_bf16(f.y) << 16);
        unsigned int hi = (unsigned int)f32_bf16(f.z) | ((unsigned int)f32_bf16(f.w) << 16);
        ((uint2*)out)[i] = make_uint2(lo, hi);
    }
}

// ---------------------------------------------------------------------------
// Tiled GEMM: C[M,F] = A[M,1024] * W[F,1024]^T + bias
// MODE 0: f32 store to out[M*F]          (output projection)
// MODE 1: bf16 scatter to q/k/v[B,H,N,D] (QKV projection, f = h*192 + d*3 + w)
// Block: 256 threads (8 waves, 4x2), tile 128x128, K-step 32.
// Double-buffered async global->LDS staging: 4 async b128 per wave per stage.
// ---------------------------------------------------------------------------
template <int MODE>
__global__ __launch_bounds__(256) void gemm_bf16_kernel(
    const unsigned short* __restrict__ A,
    const unsigned short* __restrict__ W,
    const float* __restrict__ bias,
    float* __restrict__ out,
    unsigned short* __restrict__ qv,
    unsigned short* __restrict__ kv,
    unsigned short* __restrict__ vv,
    int Fdim) {
    constexpr int KD = 1024;
    __shared__ alignas(16) unsigned short ldsA[2 * 128 * 32];
    __shared__ alignas(16) unsigned short ldsW[2 * 128 * 32];

    const int tid  = threadIdx.x;
    const int wave = tid >> 5;
    const int lane = tid & 31;
    const int half = lane >> 4;
    const int ln16 = lane & 15;
    const int wm = wave & 3;   // 32-row slab
    const int wn = wave >> 2;  // 64-col slab
    const int Mbase = blockIdx.y * 128;
    const int Fbase = blockIdx.x * 128;

    const uint64_t baseA = (uint64_t)A;
    const uint64_t baseW = (uint64_t)W;
    const uint32_t ldsA0 = lds_offset(ldsA);
    const uint32_t ldsW0 = lds_offset(ldsW);

    v8f acc[2][4];
    for (int mi = 0; mi < 2; ++mi)
        for (int ni = 0; ni < 4; ++ni) acc[mi][ni] = v8f_zero();

    // stage 128x32 bf16 slabs of A and W into LDS buffer `buf` (async)
    auto stage = [&](int k0, int buf) {
        for (int i = tid; i < 512; i += 256) {        // 2 iterations / thread
            int r = i >> 2, c = (i & 3) * 8;
            uint32_t l = (uint32_t)(buf * 8192 + i * 16);
            async_copy_b128(ldsA0 + l, baseA,
                            (uint32_t)(((Mbase + r) * KD + k0 + c) * 2));
            async_copy_b128(ldsW0 + l, baseW,
                            (uint32_t)(((Fbase + r) * KD + k0 + c) * 2));
        }
    };

    stage(0, 0);
    for (int k0 = 0, it = 0; k0 < KD; k0 += 32, ++it) {
        const int buf = it & 1;
        if (k0 + 32 < KD) {
            stage(k0 + 32, buf ^ 1);   // 4 async ops in flight for next stage
            wait_async_le4();          // current stage complete
        } else {
            wait_async_0();
        }
        __syncthreads();               // all waves' current stage visible

        const unsigned short* ldsAb = ldsA + buf * 4096;
        const unsigned short* ldsWb = ldsW + buf * 4096;
        FragBF afr[2], bfr[4];
        for (int mi = 0; mi < 2; ++mi) {
            const unsigned short* base = ldsAb + (wm * 32 + mi * 16 + ln16) * 32;
            afr[mi].q[0] = *(const uint4*)(base + half * 8);       // k = half*8 .. +7
            afr[mi].q[1] = *(const uint4*)(base + 16 + half * 8);  // k = 16+half*8 .. +7
        }
        for (int ni = 0; ni < 4; ++ni) {
            const unsigned short* base = ldsWb + (wn * 64 + ni * 16 + ln16) * 32;
            bfr[ni].q[0] = *(const uint4*)(base + half * 8);       // k = half*8 .. +7
            bfr[ni].q[1] = *(const uint4*)(base + 16 + half * 8);  // k = 16+half*8 .. +7
        }
        for (int mi = 0; mi < 2; ++mi)
            for (int ni = 0; ni < 4; ++ni)
                acc[mi][ni] = wmma_bf16(afr[mi], bfr[ni], acc[mi][ni]);
        __syncthreads();               // finished reading buf before it is rewritten
    }

    // epilogue (C layout: lane = col n, rows half*8 + r)
    for (int mi = 0; mi < 2; ++mi) {
        int rowb = Mbase + wm * 32 + mi * 16 + half * 8;
        for (int ni = 0; ni < 4; ++ni) {
            int col = Fbase + wn * 64 + ni * 16 + ln16;
            float bsv = bias[col];
            if (MODE == 0) {
                for (int r = 0; r < 8; ++r)
                    out[(size_t)(rowb + r) * Fdim + col] = acc[mi][ni][r] + bsv;
            } else {
                int wsel = col % 3;
                int d    = (col / 3) & 63;
                int h    = col / 192;
                unsigned short* dst = (wsel == 0) ? qv : (wsel == 1) ? kv : vv;
                for (int r = 0; r < 8; ++r) {
                    int row  = rowb + r;
                    int b    = row >> 11;          // N = 2048
                    int nseq = row & 2047;
                    size_t idx = (((size_t)(b * 16 + h)) * 2048 + nseq) * 64 + d;
                    dst[idx] = f32_bf16(acc[mi][ni][r] + bsv);
                }
            }
        }
    }
}

// ---------------------------------------------------------------------------
// Flash attention: one block per (bh, 128 q rows); each wave owns 16 q rows.
// S = Q K^T (raw), online softmax, O = P V; output scaled by 1/sqrt(E)=1/32,
// stored bf16 into [B, N, E] with heads concatenated.
// K blocks staged async global->LDS; V transposed through VGPRs into LDS.
// ---------------------------------------------------------------------------
__global__ __launch_bounds__(256) void attn_kernel(
    const unsigned short* __restrict__ Q,
    const unsigned short* __restrict__ K,
    const unsigned short* __restrict__ V,
    unsigned short* __restrict__ O) {
    constexpr int N = 2048, D = 64;
    __shared__ alignas(16) unsigned short ldsK[32 * 64];     // [key][d]
    __shared__ alignas(16) unsigned short ldsVt[64 * 32];    // [d][key]
    __shared__ alignas(16) unsigned short ldsP[8 * 16 * 32]; // per-wave P tile

    const int tid  = threadIdx.x;
    const int wave = tid >> 5;
    const int lane = tid & 31;
    const int half = lane >> 4;
    const int ln16 = lane & 15;
    const int bh = blockIdx.y;         // 0..63
    const int b = bh >> 4, h = bh & 15;
    const int qb = blockIdx.x * 128 + wave * 16;
    const size_t headbase = (size_t)bh * N * D;

    const uint64_t baseK = (uint64_t)K;
    const uint32_t ldsK0 = lds_offset(ldsK);

    // Q fragments for this wave's 16 rows (A-layout, two 32-wide d chunks)
    FragBF aq[2];
    {
        const unsigned short* qrow = Q + headbase + (size_t)(qb + ln16) * D;
        aq[0].q[0] = *(const uint4*)(qrow + half * 8);
        aq[0].q[1] = *(const uint4*)(qrow + 16 + half * 8);
        aq[1].q[0] = *(const uint4*)(qrow + 32 + half * 8);
        aq[1].q[1] = *(const uint4*)(qrow + 48 + half * 8);
    }

    v8f o[4];
    for (int t = 0; t < 4; ++t) o[t] = v8f_zero();
    float mrun[8], lrun[8];
    for (int r = 0; r < 8; ++r) { mrun[r] = -3.0e38f; lrun[r] = 0.0f; }

    const int svRow = tid >> 3;        // 0..31 key row
    const int svCol = (tid & 7) * 8;   // d offset

    for (int j = 0; j < N; j += 32) {
        // stage K block [32,64] async (1 b128 per thread = 1 async op per wave)
        async_copy_b128(ldsK0 + (uint32_t)(tid * 16), baseK,
                        (uint32_t)((headbase + (size_t)(j + svRow) * D + svCol) * 2));
        // stage V^T block [64,32] through VGPRs (scatter needs registers anyway)
        {
            uint4 vq = *(const uint4*)(V + headbase + (size_t)(j + svRow) * D + svCol);
            const unsigned short* vp = (const unsigned short*)&vq;
            for (int i = 0; i < 8; ++i) ldsVt[(svCol + i) * 32 + svRow] = vp[i];
        }
        wait_async_0();
        __syncthreads();

        // S tiles: 2 key tiles x (2 wmma over d)
        v8f s[2];
        for (int kt = 0; kt < 2; ++kt) {
            FragBF bk0, bk1;
            const unsigned short* kbase = ldsK + (kt * 16 + ln16) * 64;
            bk0.q[0] = *(const uint4*)(kbase + half * 8);
            bk0.q[1] = *(const uint4*)(kbase + 16 + half * 8);
            bk1.q[0] = *(const uint4*)(kbase + 32 + half * 8);
            bk1.q[1] = *(const uint4*)(kbase + 48 + half * 8);
            v8f z = v8f_zero();
            z = wmma_bf16(aq[0], bk0, z);
            z = wmma_bf16(aq[1], bk1, z);
            s[kt] = z;
        }

        // online softmax per owned row (row m = half*8+r spans 16 lanes)
        float p0s[8], p1s[8];
        for (int r = 0; r < 8; ++r) {
            float s0r = s[0][r], s1r = s[1][r];
            float mx = fmaxf(s0r, s1r);
            mx = fmaxf(mx, __shfl_xor(mx, 1, 32));
            mx = fmaxf(mx, __shfl_xor(mx, 2, 32));
            mx = fmaxf(mx, __shfl_xor(mx, 4, 32));
            mx = fmaxf(mx, __shfl_xor(mx, 8, 32));
            float mnew = fmaxf(mrun[r], mx);
            float corr = __expf(mrun[r] - mnew);
            float p0 = __expf(s0r - mnew);
            float p1 = __expf(s1r - mnew);
            float rs = p0 + p1;
            rs += __shfl_xor(rs, 1, 32);
            rs += __shfl_xor(rs, 2, 32);
            rs += __shfl_xor(rs, 4, 32);
            rs += __shfl_xor(rs, 8, 32);
            lrun[r] = lrun[r] * corr + rs;
            mrun[r] = mnew;
            for (int t = 0; t < 4; ++t) o[t][r] *= corr;
            p0s[r] = p0; p1s[r] = p1;
        }

        // transpose P through per-wave LDS into A-fragment layout
        {
            unsigned short* pw = ldsP + wave * 512;
            for (int r = 0; r < 8; ++r) {
                int m = half * 8 + r;
                pw[m * 32 + ln16]      = f32_bf16(p0s[r]);
                pw[m * 32 + 16 + ln16] = f32_bf16(p1s[r]);
            }
        }
        __syncthreads();

        FragBF pf;
        {
            const unsigned short* pr = ldsP + wave * 512 + ln16 * 32;
            pf.q[0] = *(const uint4*)(pr + half * 8);
            pf.q[1] = *(const uint4*)(pr + 16 + half * 8);
        }
        for (int t = 0; t < 4; ++t) {
            FragBF bv;
            const unsigned short* vb = ldsVt + (t * 16 + ln16) * 32;
            bv.q[0] = *(const uint4*)(vb + half * 8);
            bv.q[1] = *(const uint4*)(vb + 16 + half * 8);
            o[t] = wmma_bf16(pf, bv, o[t]);
        }
        __syncthreads();               // protect ldsK/ldsVt before next stage
    }

    // normalize (1/l) * 1/sqrt(E) and store bf16 to [B,N,E]
    const float headscale = 0.03125f;  // 1/sqrt(1024)
    for (int r = 0; r < 8; ++r) {
        float inv = headscale / lrun[r];
        int row = qb + half * 8 + r;
        unsigned short* orow = O + ((size_t)(b * N + row)) * 1024 + h * 64;
        for (int t = 0; t < 4; ++t)
            orow[t * 16 + ln16] = f32_bf16(o[t][r] * inv);
    }
}

// ---------------------------------------------------------------------------
// Host side
// ---------------------------------------------------------------------------
extern "C" void kernel_launch(void* const* d_in, const int* in_sizes, int n_in,
                              void* d_out, int out_size, void* d_ws, size_t ws_size,
                              hipStream_t stream) {
    (void)in_sizes; (void)n_in; (void)out_size; (void)ws_size;
    const float* x     = (const float*)d_in[0];  // [4,2048,1024]
    const float* Wqkv  = (const float*)d_in[1];  // [3072,1024]
    const float* bqkv  = (const float*)d_in[2];  // [3072]
    const float* Wproj = (const float*)d_in[3];  // [1024,1024]
    const float* bproj = (const float*)d_in[4];  // [1024]
    float* out = (float*)d_out;                  // [4,2048,1024]

    char* p = (char*)d_ws;
    auto carve = [&](size_t bytes) -> unsigned short* {
        unsigned short* r = (unsigned short*)p;
        p += (bytes + 255) & ~(size_t)255;
        return r;
    };
    unsigned short* xb     = carve(8192ull * 1024 * 2);  // x bf16
    unsigned short* wqkvb  = carve(3072ull * 1024 * 2);  // W_qkv bf16
    unsigned short* wprojb = carve(1024ull * 1024 * 2);  // W_proj bf16
    unsigned short* qb     = carve(64ull * 2048 * 64 * 2);
    unsigned short* kb     = carve(64ull * 2048 * 64 * 2);
    unsigned short* vb     = carve(64ull * 2048 * 64 * 2);
    unsigned short* attb   = carve(8192ull * 1024 * 2);  // attention out bf16

    // 1) bf16 conversions
    {
        int n4 = (8192 * 1024) / 4;
        cvt_f32_bf16_kernel<<<n4 / 256, 256, 0, stream>>>(x, xb, n4);
        n4 = (3072 * 1024) / 4;
        cvt_f32_bf16_kernel<<<n4 / 256, 256, 0, stream>>>(Wqkv, wqkvb, n4);
        n4 = (1024 * 1024) / 4;
        cvt_f32_bf16_kernel<<<n4 / 256, 256, 0, stream>>>(Wproj, wprojb, n4);
    }
    // 2) QKV projection -> q/k/v bf16 [B,H,N,D]
    gemm_bf16_kernel<1><<<dim3(3072 / 128, 8192 / 128), 256, 0, stream>>>(
        xb, wqkvb, bqkv, nullptr, qb, kb, vb, 3072);
    // 3) flash attention -> attb bf16 [B,N,E]
    attn_kernel<<<dim3(2048 / 128, 64), 256, 0, stream>>>(qb, kb, vb, attb);
    // 4) output projection -> d_out f32
    gemm_bf16_kernel<0><<<dim3(1024 / 128, 8192 / 128), 256, 0, stream>>>(
        attb, wprojb, bproj, out, nullptr, nullptr, nullptr, 1024);
}
